// LTX2Attention_41850161332482
// MI455X (gfx1250) — compile-verified, matrix operand order checked by reading, not run on hardware
//
#include <hip/hip_runtime.h>
#include <hip/hip_bf16.h>

typedef __attribute__((ext_vector_type(16))) _Float16 v16h;
typedef __attribute__((ext_vector_type(8)))  _Float16 v8h;
typedef __attribute__((ext_vector_type(4)))  _Float16 v4h;
typedef __attribute__((ext_vector_type(8)))  float    v8f;
typedef __attribute__((ext_vector_type(4)))  unsigned int v4u;
typedef __attribute__((ext_vector_type(8)))  int      v8i;
typedef __attribute__((ext_vector_type(4)))  int      v4i;

#define WMMA_F16(a, b, c) \
  __builtin_amdgcn_wmma_f32_16x16x32_f16(false, (a), false, (b), (short)0, (c), false, false)

__device__ __forceinline__ unsigned lds_addr(const void* p) {
  return (unsigned)(uintptr_t)p;  // low 32 bits of generic shared ptr = LDS offset
}

// ---------------------------------------------------------------------------
// Tensor Data Mover: one-instruction 2D tile DMA, global -> LDS (TENSORcnt).
// D# per CDNA5 ISA ch.8: data_size=3 (8-byte units); tensor dims == tile dims
// (tile anchored at global_addr, no OOB); tensor_dim0_stride = row pitch.
//   rowUnits   : tile row length in 8B units
//   rows       : number of tile rows
//   strideUnits: pitch between tensor rows in 8B units
// ---------------------------------------------------------------------------
__device__ __forceinline__ void tdm_load_2d(const void* gaddr, unsigned ldsaddr,
                                            unsigned rowUnits, unsigned rows,
                                            unsigned strideUnits) {
  const unsigned long long ga = (unsigned long long)(uintptr_t)gaddr;
  v4u g0;
  g0[0] = 1u;                                                 // count=1 (valid D#)
  g0[1] = ldsaddr;                                            // lds_addr [63:32]
  g0[2] = (unsigned)(ga & 0xFFFFFFFFu);                       // global_addr[31:0]
  g0[3] = (unsigned)((ga >> 32) & 0x1FFFFFFu) | (2u << 30);   // ga[56:32] | type=2
  v8i g1;
  g1[0] = (int)(3u << 16);                    // workgroup_mask=0 | data_size=3 (8B)
  g1[1] = (int)((rowUnits & 0xFFFFu) << 16);  // tensor_dim0[15:0]  (bits 79:48)
  g1[2] = (int)((rows & 0xFFFFu) << 16);      // td0 hi=0 | tensor_dim1[15:0]
  g1[3] = (int)((rowUnits & 0xFFFFu) << 16);  // td1 hi=0 | tile_dim0 (bits 127:112)
  g1[4] = (int)(rows & 0xFFFFu);              // tile_dim1 | tile_dim2=0
  g1[5] = (int)strideUnits;                   // tensor_dim0_stride[31:0]
  g1[6] = 0;                                  // stride hi | tensor_dim1_stride lo
  g1[7] = 0;
  const v4i z4 = {0, 0, 0, 0};
#if __clang_major__ >= 23
  const v8i z8 = {0, 0, 0, 0, 0, 0, 0, 0};
  __builtin_amdgcn_tensor_load_to_lds(g0, g1, z4, z4, z8, 0);
#else
  __builtin_amdgcn_tensor_load_to_lds(g0, g1, z4, z4, 0);
#endif
}

// ---------------------------------------------------------------------------
// GEMM:  C[M,N] (f32) = A[M,K](f16) x W[N,K](f16)^T + bias[N]
// 256 threads = 8 waves arranged 4x2; workgroup tile 128x128; K-step 64.
// Double-buffered LDS; each K-tile (A 128x128B rows + W 128x128B rows) is
// fetched by TWO tensor_load_to_lds DMAs issued by wave 0 — tile i+1 streams
// via the TDM while tile i feeds the WMMA pipe.
// ---------------------------------------------------------------------------
__global__ __launch_bounds__(256) void gemm_f16_kernel(
    const _Float16* __restrict__ Ah, const _Float16* __restrict__ Wh,
    const float* __restrict__ bias, float* __restrict__ C,
    int M, int N, int K)
{
  __shared__ _Float16 As[2][128 * 64];  // 2 x 16KB
  __shared__ _Float16 Bs[2][128 * 64];  // 2 x 16KB

  const int tid  = threadIdx.x;
  const int lane = tid & 31;
  const int wave = tid >> 5;
  const int ln   = lane & 15;
  const int hf   = lane >> 4;
  const int rowg = wave >> 1;  // 0..3 : 32-row group
  const int colg = wave & 1;   // 0..1 : 64-col group

  const int n0 = blockIdx.x * 128;
  const int m0 = blockIdx.y * 128;

  const unsigned strideU = (unsigned)(K / 4);  // K f16 = K*2 B = K/4 8B-units
  const _Float16* aBase = Ah + (size_t)m0 * K;
  const _Float16* bBase = Wh + (size_t)n0 * K;
  unsigned aL[2] = { lds_addr(&As[0][0]), lds_addr(&As[1][0]) };
  unsigned bL[2] = { lds_addr(&Bs[0][0]), lds_addr(&Bs[1][0]) };

  auto issueTile = [&](int buf, int k0) {
    tdm_load_2d(aBase + k0, aL[buf], 16, 128, strideU);  // A tile 128 x 128B
    tdm_load_2d(bBase + k0, bL[buf], 16, 128, strideU);  // W tile 128 x 128B
  };

  v8f acc[2][4] = {};

  const int NT = K / 64;
  if (wave == 0) issueTile(0, 0);
  for (int it = 0; it < NT; ++it) {
    const int buf = it & 1;
    if (it + 1 < NT) {
      if (wave == 0) { issueTile(buf ^ 1, (it + 1) * 64); __builtin_amdgcn_s_wait_tensorcnt(2); }
    } else {
      if (wave == 0) __builtin_amdgcn_s_wait_tensorcnt(0);
    }
    __syncthreads();  // tile `it` visible to all waves

#pragma unroll
    for (int kc = 0; kc < 2; kc++) {
      v16h a0 = *(const v16h*)&As[buf][(rowg * 32 + ln) * 64 + kc * 32 + hf * 16];
      v16h a1 = *(const v16h*)&As[buf][(rowg * 32 + 16 + ln) * 64 + kc * 32 + hf * 16];
#pragma unroll
      for (int nb = 0; nb < 4; nb++) {
        v16h b = *(const v16h*)&Bs[buf][(colg * 64 + nb * 16 + ln) * 64 + kc * 32 + hf * 16];
        acc[0][nb] = WMMA_F16(a0, b, acc[0][nb]);
        acc[1][nb] = WMMA_F16(a1, b, acc[1][nb]);
      }
    }
    __syncthreads();  // all waves done with buf before the TDM refills it
  }

  // ---- epilogue: bias + f32 store ----
#pragma unroll
  for (int r = 0; r < 2; r++) {
#pragma unroll
    for (int nb = 0; nb < 4; nb++) {
      const int col = n0 + colg * 64 + nb * 16 + ln;
      const float bb = bias[col];
#pragma unroll
      for (int j = 0; j < 8; j++) {
        const int row = m0 + rowg * 32 + r * 16 + j + 8 * hf;
        C[(size_t)row * N + col] = acc[r][nb][j] + bb;
      }
    }
  }
}

// ---------------------------------------------------------------------------
// Fused RMSNorm (over D) + interleaved RoPE + f16 downconvert.
// ---------------------------------------------------------------------------
__global__ __launch_bounds__(256) void rmsnorm_rope_kernel(
    const float* __restrict__ src, const float* __restrict__ w,
    const float* __restrict__ cosb, const float* __restrict__ sinb,
    _Float16* __restrict__ dst, int T, int D, float outScale)
{
  __shared__ float red[8];
  const int row = blockIdx.x;  // b*T + t
  const int t   = row % T;
  const float* h = src + (size_t)row * D;
  const int base = threadIdx.x * 8;

  float4 h0 = *(const float4*)&h[base];
  float4 h1 = *(const float4*)&h[base + 4];
  float ss = h0.x * h0.x + h0.y * h0.y + h0.z * h0.z + h0.w * h0.w +
             h1.x * h1.x + h1.y * h1.y + h1.z * h1.z + h1.w * h1.w;
#pragma unroll
  for (int off = 16; off > 0; off >>= 1) ss += __shfl_xor(ss, off, 32);
  if ((threadIdx.x & 31) == 0) red[threadIdx.x >> 5] = ss;
  __syncthreads();
  float tot = 0.f;
#pragma unroll
  for (int i = 0; i < 8; i++) tot += red[i];
  const float rinv = rsqrtf(tot / (float)D + 1e-6f);

  const float hv[8] = {h0.x, h0.y, h0.z, h0.w, h1.x, h1.y, h1.z, h1.w};
  const float* cb = cosb + (size_t)t * D + base;
  const float* sb = sinb + (size_t)t * D + base;
  const float* wb = w + base;
  _Float16* dp = dst + (size_t)row * D + base;
#pragma unroll
  for (int p = 0; p < 4; p++) {
    const int e = 2 * p, o = 2 * p + 1;
    const float he = hv[e] * wb[e] * rinv;
    const float ho = hv[o] * wb[o] * rinv;
    dp[e] = (_Float16)((he * cb[e] - ho * sb[e]) * outScale);
    dp[o] = (_Float16)((ho * cb[o] + he * sb[o]) * outScale);
  }
}

// f32 -> f16 bulk convert (x and W matrices)
__global__ __launch_bounds__(256) void cvt_f16_kernel(
    const float* __restrict__ src, _Float16* __restrict__ dst, int n4)
{
  const int i = blockIdx.x * 256 + threadIdx.x;
  if (i >= n4) return;
  float4 f = ((const float4*)src)[i];
  v4h o;
  o[0] = (_Float16)f.x; o[1] = (_Float16)f.y;
  o[2] = (_Float16)f.z; o[3] = (_Float16)f.w;
  ((v4h*)dst)[i] = o;
}

// ---------------------------------------------------------------------------
// V transpose: f32 [b, t, h*128+dh]  ->  f16 [b][h][dh][t]  (LDS-tiled)
// so the attention kernel can TDM-load V tiles with contiguous keys.
// ---------------------------------------------------------------------------
__global__ __launch_bounds__(256) void transpose_v_kernel(
    const float* __restrict__ vf, _Float16* __restrict__ vT, int T)
{
  __shared__ _Float16 Ls[64 * 128];  // [t][dh] f16, 16KB
  const int tid = threadIdx.x;
  const int t0 = blockIdx.x * 64;
  const int hh = blockIdx.y, b = blockIdx.z;
  const int D = 2048, H = 16, Dh = 128;

  {  // load + convert 64t x 128dh tile (coalesced reads)
    const int r = tid >> 2, q = tid & 3;
    const float* sp = vf + (size_t)(b * T + t0 + r) * D + hh * Dh + q * 32;
    v16h u0, u1;
#pragma unroll
    for (int i = 0; i < 16; i++) { u0[i] = (_Float16)sp[i]; u1[i] = (_Float16)sp[16 + i]; }
    *(v16h*)&Ls[r * 128 + q * 32]      = u0;
    *(v16h*)&Ls[r * 128 + q * 32 + 16] = u1;
  }
  __syncthreads();
  {  // write transposed (coalesced writes over t)
    const int dh = tid >> 1, half = tid & 1;
    v16h o0, o1;
#pragma unroll
    for (int i = 0; i < 16; i++) {
      o0[i] = Ls[(half * 32 + i) * 128 + dh];
      o1[i] = Ls[(half * 32 + 16 + i) * 128 + dh];
    }
    _Float16* dp = vT + ((size_t)(b * H + hh) * Dh + dh) * T + t0 + half * 32;
    *(v16h*)dp        = o0;
    *(v16h*)(dp + 16) = o1;
  }
}

// ---------------------------------------------------------------------------
// Flash attention, non-causal. Workgroup = (b, h, 128 q rows); 8 waves x 16
// rows. K tiles (double-buffered) and V tiles streamed by the Tensor Data
// Mover: wave 0 issues one tensor_load_to_lds per tile; in-order pending
// queue is [K_i, V_i, K_i+1] so s_wait_tensorcnt 1 drains the current tile.
// Q pre-scaled by 1/sqrt(Dh); online softmax in f32; P transposed through
// per-wave LDS scratch.
// ---------------------------------------------------------------------------
__global__ __launch_bounds__(256) void flash_attn_kernel(
    const _Float16* __restrict__ qh, const _Float16* __restrict__ kh,
    const _Float16* __restrict__ vT, _Float16* __restrict__ oh, int T)
{
  __shared__ _Float16 Kt[2][64 * 128];  // [key][dh], 2 x 16KB
  __shared__ _Float16 Vt[128 * 64];     // [dh][key], 16KB
  __shared__ _Float16 Ps[8][16 * 64];   // per-wave P scratch [m][key], 16KB

  const int tid  = threadIdx.x;
  const int lane = tid & 31;
  const int wave = tid >> 5;
  const int ln   = lane & 15;
  const int hf   = lane >> 4;

  const int q0 = blockIdx.x * 128;
  const int hh = blockIdx.y;
  const int b  = blockIdx.z;
  const int D = 2048, Dh = 128, H = 16;

  // Q A-fragments: 4 chunks of K=32 covering Dh=128 (kept resident)
  const size_t qrow = (size_t)(b * T + q0 + wave * 16 + ln);
  v16h qA[4];
#pragma unroll
  for (int kc = 0; kc < 4; kc++)
    qA[kc] = *(const v16h*)&qh[qrow * D + hh * Dh + kc * 32 + hf * 16];

  v8f oAcc[8] = {};
  float rowm[8], rowl[8];
#pragma unroll
  for (int j = 0; j < 8; j++) { rowm[j] = -3.0e38f; rowl[j] = 0.f; }

  const _Float16* kBase = kh + (size_t)(b * T) * D + hh * Dh;
  const _Float16* vBase = vT + (size_t)(b * H + hh) * Dh * T;
  unsigned kL[2] = { lds_addr(&Kt[0][0]), lds_addr(&Kt[1][0]) };
  const unsigned vLb = lds_addr(&Vt[0]);

  auto tdmK = [&](int buf, int kt0) {  // 64 keys x 256B rows, pitch D
    tdm_load_2d(kBase + (size_t)kt0 * D, kL[buf], 32, 64, (unsigned)(D / 4));
  };
  auto tdmV = [&](int kt0) {           // 128 dh x 128B rows, pitch T
    tdm_load_2d(vBase + kt0, vLb, 16, 128, (unsigned)(T / 4));
  };

  const int NT = T / 64;
  if (wave == 0) tdmK(0, 0);
  for (int it = 0; it < NT; ++it) {
    const int kt0 = it * 64;
    const int buf = it & 1;
    if (wave == 0) {
      tdmV(kt0);
      if (it + 1 < NT) { tdmK(buf ^ 1, kt0 + 64); __builtin_amdgcn_s_wait_tensorcnt(1); }
      else             { __builtin_amdgcn_s_wait_tensorcnt(0); }
    }
    __syncthreads();

    // ---- S = Q * K^T (16 WMMA), scores for 16 q x 64 keys ----
    v8f s[4] = {};
#pragma unroll
    for (int kc = 0; kc < 4; kc++) {
#pragma unroll
      for (int n = 0; n < 4; n++) {
        v16h kb = *(const v16h*)&Kt[buf][(n * 16 + ln) * 128 + kc * 32 + hf * 16];
        s[n] = WMMA_F16(qA[kc], kb, s[n]);
      }
    }

    // ---- online softmax (row stats per M = j + 8*hf) ----
#pragma unroll
    for (int j = 0; j < 8; j++) {
      float t = fmaxf(fmaxf(s[0][j], s[1][j]), fmaxf(s[2][j], s[3][j]));
#pragma unroll
      for (int off = 1; off < 16; off <<= 1) t = fmaxf(t, __shfl_xor(t, off, 32));
      const float mnew  = fmaxf(rowm[j], t);
      const float alpha = __expf(rowm[j] - mnew);
      rowm[j] = mnew;
      rowl[j] *= alpha;
#pragma unroll
      for (int nd = 0; nd < 8; nd++) oAcc[nd][j] *= alpha;
      float part = 0.f;
#pragma unroll
      for (int n = 0; n < 4; n++) {
        const float p = __expf(s[n][j] - mnew);
        s[n][j] = p;
        part += p;
      }
#pragma unroll
      for (int off = 1; off < 16; off <<= 1) part += __shfl_xor(part, off, 32);
      rowl[j] += part;
    }

    // ---- transpose P (C layout -> A layout) through per-wave LDS ----
#pragma unroll
    for (int j = 0; j < 8; j++)
#pragma unroll
      for (int n = 0; n < 4; n++)
        Ps[wave][(j + 8 * hf) * 64 + n * 16 + ln] = (_Float16)s[n][j];
    asm volatile("s_wait_dscnt 0" ::: "memory");  // wave-local LDS RAW fence

    v16h pA0 = *(const v16h*)&Ps[wave][ln * 64 + 0 * 32 + hf * 16];
    v16h pA1 = *(const v16h*)&Ps[wave][ln * 64 + 1 * 32 + hf * 16];

    // ---- O += P * V (16 WMMA) ----
#pragma unroll
    for (int kc = 0; kc < 2; kc++) {
      const v16h pa = kc ? pA1 : pA0;
#pragma unroll
      for (int nd = 0; nd < 8; nd++) {
        v16h vb = *(const v16h*)&Vt[(nd * 16 + ln) * 64 + kc * 32 + hf * 16];
        oAcc[nd] = WMMA_F16(pa, vb, oAcc[nd]);
      }
    }
    __syncthreads();  // everyone done with Kt[buf]/Vt before the TDM refills
  }

  // ---- normalize and store f16 attention output ----
#pragma unroll
  for (int nd = 0; nd < 8; nd++) {
#pragma unroll
    for (int j = 0; j < 8; j++) {
      const size_t row = (size_t)(b * T + q0 + wave * 16 + j + 8 * hf);
      oh[row * D + hh * Dh + nd * 16 + ln] = (_Float16)(oAcc[nd][j] / rowl[j]);
    }
  }
}

// ---------------------------------------------------------------------------
extern "C" void kernel_launch(void* const* d_in, const int* in_sizes, int n_in,
                              void* d_out, int out_size, void* d_ws, size_t ws_size,
                              hipStream_t stream)
{
  (void)in_sizes; (void)n_in; (void)out_size; (void)ws_size;
  const float* x    = (const float*)d_in[0];
  const float* cosb = (const float*)d_in[1];
  const float* sinb = (const float*)d_in[2];
  const float* Wq   = (const float*)d_in[3];
  const float* bq   = (const float*)d_in[4];
  const float* Wk   = (const float*)d_in[5];
  const float* bk   = (const float*)d_in[6];
  const float* Wv   = (const float*)d_in[7];
  const float* bv   = (const float*)d_in[8];
  const float* qnw  = (const float*)d_in[9];
  const float* knw  = (const float*)d_in[10];
  const float* Wo   = (const float*)d_in[11];
  const float* bo   = (const float*)d_in[12];
  float* out = (float*)d_out;

  const int B = 2, T = 2048, D = 2048, H = 16;
  const int M = B * T;

  // Workspace layout (208 MB total)
  char* ws = (char*)d_ws;
  const size_t MB = 1ull << 20;
  float*    qf  = (float*)(ws + 0 * MB);       // 32 MB f32 q
  float*    kf  = (float*)(ws + 32 * MB);      // 32 MB f32 k
  float*    vf  = (float*)(ws + 64 * MB);      // 32 MB f32 v
  _Float16* qh  = (_Float16*)(ws + 96 * MB);   // 16 MB f16 q-hat
  _Float16* kh  = (_Float16*)(ws + 112 * MB);  // 16 MB f16 k-hat
  _Float16* vTh = (_Float16*)(ws + 128 * MB);  // 16 MB f16 V transposed
  _Float16* oh  = (_Float16*)(ws + 144 * MB);  // 16 MB f16 attn out
  _Float16* xh  = (_Float16*)(ws + 160 * MB);  // 16 MB f16 x
  _Float16* Wqh = (_Float16*)(ws + 176 * MB);  //  8 MB
  _Float16* Wkh = (_Float16*)(ws + 184 * MB);  //  8 MB
  _Float16* Wvh = (_Float16*)(ws + 192 * MB);  //  8 MB
  _Float16* Woh = (_Float16*)(ws + 200 * MB);  //  8 MB

  dim3 blk(256);
  const int nx4 = M * D / 4, nw4 = D * D / 4;
  cvt_f16_kernel<<<(nx4 + 255) / 256, blk, 0, stream>>>(x,  xh,  nx4);
  cvt_f16_kernel<<<(nw4 + 255) / 256, blk, 0, stream>>>(Wq, Wqh, nw4);
  cvt_f16_kernel<<<(nw4 + 255) / 256, blk, 0, stream>>>(Wk, Wkh, nw4);
  cvt_f16_kernel<<<(nw4 + 255) / 256, blk, 0, stream>>>(Wv, Wvh, nw4);
  cvt_f16_kernel<<<(nw4 + 255) / 256, blk, 0, stream>>>(Wo, Woh, nw4);

  dim3 gg(D / 128, M / 128);
  gemm_f16_kernel<<<gg, blk, 0, stream>>>(xh, Wqh, bq, qf, M, D, D);
  gemm_f16_kernel<<<gg, blk, 0, stream>>>(xh, Wkh, bk, kf, M, D, D);
  gemm_f16_kernel<<<gg, blk, 0, stream>>>(xh, Wvh, bv, vf, M, D, D);

  const float qscale = 0.08838834764831845f;  // 1/sqrt(128), folded into Q
  rmsnorm_rope_kernel<<<M, blk, 0, stream>>>(qf, qnw, cosb, sinb, qh, T, D, qscale);
  rmsnorm_rope_kernel<<<M, blk, 0, stream>>>(kf, knw, cosb, sinb, kh, T, D, 1.0f);
  transpose_v_kernel<<<dim3(T / 64, H, B), blk, 0, stream>>>(vf, vTh, T);

  flash_attn_kernel<<<dim3(T / 128, H, B), blk, 0, stream>>>(qh, kh, vTh, oh, T);

  gemm_f16_kernel<<<gg, blk, 0, stream>>>(oh, Woh, bo, out, M, D, D);
}
